// MultiHeadAttentionBlock_4320737100126
// MI455X (gfx1250) — matrix operand order repaired and checked by reading, not compiled, not run on hardware
//
#include <hip/hip_runtime.h>

// ---------- types ----------
typedef __attribute__((ext_vector_type(16))) _Float16 v16h;
typedef __attribute__((ext_vector_type(8)))  _Float16 v8h;
typedef __attribute__((ext_vector_type(8)))  float    v8f;

#define N_TOK 4096
#define CH    256
#define NH    4
#define HD    64

static __device__ __forceinline__ v8f wmma_f16(v16h a, v16h b, v8f c) {
    // v_wmma_f32_16x16x32_f16: D = A(16x32) * B(32x16) + C(16x16 f32)
    return __builtin_amdgcn_wmma_f32_16x16x32_f16(
        false, a, false, b, (short)0, c, false, false);
}

// A-fragment (16-bit, 16x32): lane = row (l&15), h = l>>4.
// elements 0..7  -> K = 16h + e      (contiguous 16B)
// elements 8..15 -> K = 16h + 8 + e  (contiguous 16B)
static __device__ __forceinline__ v16h load_a_f16(const _Float16* row_kbase, int h) {
    v8h lo = *(const v8h*)(row_kbase + 16 * h);
    v8h hi = *(const v8h*)(row_kbase + 16 * h + 8);
    v16h a;
#pragma unroll
    for (int i = 0; i < 8; ++i) { a[i] = lo[i]; a[i + 8] = hi[i]; }
    return a;
}

// ---------------------------------------------------------------------------
// Kernel 1: QKV projection.  qkv = x^T(b) (N x C) @ w_qkv^T (C x 3C) + b_qkv
// Writes Q (scaled by 1/8), K as (b,h,n,d) f16; V transposed as (b,h,d,n) f16.
// One wave -> one 16x16 output tile; K=256 -> 8 WMMA steps.
// ---------------------------------------------------------------------------
__global__ __launch_bounds__(256) void qkv_proj_kernel(
    const float* __restrict__ x, const float* __restrict__ w_qkv,
    const float* __restrict__ b_qkv,
    _Float16* __restrict__ Qb, _Float16* __restrict__ Kb,
    _Float16* __restrict__ Vt) {
    const int lane = threadIdx.x & 31;
    const int warp = threadIdx.x >> 5;
    const int wid  = blockIdx.x * 8 + warp;

    const int col_tile = wid % 48;        // 768 / 16 columns of QKV
    const int row_tile = wid / 48;        // 1024 row tiles (B*N/16)
    const int b  = row_tile >> 8;
    const int m0 = (row_tile & 255) * 16;

    const int h  = lane >> 4;
    const int nn = lane & 15;
    const int t  = m0 + nn;               // A-fragment row = token
    const int o  = col_tile * 16 + nn;    // B-fragment column = out feature

    const float* xb = x + (size_t)b * CH * N_TOK;

    v8f acc = {};
#pragma unroll
    for (int kc = 0; kc < 8; ++kc) {
        v16h a, bf;
#pragma unroll
        for (int i = 0; i < 8; ++i) {     // A: x^T, gather along channel (stride N)
            a[i]     = (_Float16)xb[(size_t)(kc * 32 + 16 * h + i) * N_TOK + t];
            a[i + 8] = (_Float16)xb[(size_t)(kc * 32 + 16 * h + 8 + i) * N_TOK + t];
        }
        const float* wr = w_qkv + (size_t)o * CH + kc * 32 + 16 * h;
#pragma unroll
        for (int i = 0; i < 16; ++i) bf[i] = (_Float16)wr[i];   // B: contiguous
        acc = wmma_f16(a, bf, acc);
    }

    const float bias = b_qkv[o];
    const int seg  = o >> 8;              // 0=Q 1=K 2=V (uniform per wave)
    const int oc   = o & 255;
    const int head = oc >> 6;
    const int d    = oc & 63;

#pragma unroll
    for (int r = 0; r < 8; ++r) {
        const float v  = acc[r] + bias;
        const int tok  = m0 + r + 8 * h;  // C-layout: VGPR r -> row r+8h
        if (seg == 0)
            Qb[(((size_t)b * NH + head) * N_TOK + tok) * HD + d] = (_Float16)(v * 0.125f);
        else if (seg == 1)
            Kb[(((size_t)b * NH + head) * N_TOK + tok) * HD + d] = (_Float16)v;
        else
            Vt[(((size_t)b * NH + head) * HD + d) * N_TOK + tok] = (_Float16)v;
    }
}

// ---------------------------------------------------------------------------
// Kernel 2: fused flash attention. One wave owns 16 query rows, streams keys
// in blocks of 32. Scores stay in registers; P goes through a 1KB per-wave
// LDS tile for the C-layout -> A-layout transpose.
// ---------------------------------------------------------------------------
__global__ __launch_bounds__(256) void flash_attn_kernel(
    const _Float16* __restrict__ Qb, const _Float16* __restrict__ Kb,
    const _Float16* __restrict__ Vt, _Float16* __restrict__ Att) {
    __shared__ _Float16 lds[8][16][32];   // per-wave 16 rows x 32 keys

    const int lane = threadIdx.x & 31;
    const int warp = threadIdx.x >> 5;
    const int wid  = blockIdx.x * 8 + warp;

    const int qtile = wid & 255;          // 256 query tiles per (b,h)
    const int bh    = wid >> 8;           // 0..15
    const int bq    = bh >> 2;
    const int head  = bh & 3;
    const int h     = lane >> 4;
    const int n     = lane & 15;
    const int m0    = qtile * 16;

    const _Float16* Qbase = Qb + (size_t)bh * N_TOK * HD;
    const _Float16* Kbase = Kb + (size_t)bh * N_TOK * HD;
    const _Float16* Vbase = Vt + (size_t)bh * HD * N_TOK;

    // Q fragments (scale pre-folded): row = m0 + (lane&15)
    const _Float16* qrow = Qbase + (size_t)(m0 + n) * HD;
    const v16h qa0 = load_a_f16(qrow, h);        // k-dims 0..31
    const v16h qa1 = load_a_f16(qrow + 32, h);   // k-dims 32..63

    v8f o0 = {}, o1 = {}, o2 = {}, o3 = {};
    float rmax[8], rsum[8];
#pragma unroll
    for (int r = 0; r < 8; ++r) { rmax[r] = -1e30f; rsum[r] = 0.0f; }

    for (int kt = 0; kt < N_TOK; kt += 32) {
        // ---- S = Q * K^T for 32 keys (two 16x16 blocks) ----
        const _Float16* krow0 = Kbase + (size_t)(kt + n) * HD;
        const _Float16* krow1 = Kbase + (size_t)(kt + 16 + n) * HD;
        v16h kb00 = *(const v16h*)(krow0 + 16 * h);       // d 0..31
        v16h kb01 = *(const v16h*)(krow0 + 32 + 16 * h);  // d 32..63
        v16h kb10 = *(const v16h*)(krow1 + 16 * h);
        v16h kb11 = *(const v16h*)(krow1 + 32 + 16 * h);
        v8f s0 = {}, s1 = {};
        s0 = wmma_f16(qa0, kb00, s0);
        s0 = wmma_f16(qa1, kb01, s0);
        s1 = wmma_f16(qa0, kb10, s1);
        s1 = wmma_f16(qa1, kb11, s1);

        // ---- online softmax (row reductions across the 16-lane half) ----
#pragma unroll
        for (int r = 0; r < 8; ++r) {
            float v = fmaxf(s0[r], s1[r]);
            v = fmaxf(v, __shfl_xor(v, 1));
            v = fmaxf(v, __shfl_xor(v, 2));
            v = fmaxf(v, __shfl_xor(v, 4));
            v = fmaxf(v, __shfl_xor(v, 8));
            const float nm    = fmaxf(rmax[r], v);
            const float scale = __expf(rmax[r] - nm);
            const float p0    = __expf(s0[r] - nm);
            const float p1    = __expf(s1[r] - nm);
            float ps = p0 + p1;
            ps += __shfl_xor(ps, 1);
            ps += __shfl_xor(ps, 2);
            ps += __shfl_xor(ps, 4);
            ps += __shfl_xor(ps, 8);
            rsum[r] = rsum[r] * scale + ps;
            rmax[r] = nm;
            o0[r] *= scale; o1[r] *= scale; o2[r] *= scale; o3[r] *= scale;
            lds[warp][r + 8 * h][n]      = (_Float16)p0;   // transpose via LDS
            lds[warp][r + 8 * h][16 + n] = (_Float16)p1;
        }
        asm volatile("s_wait_dscnt 0" ::: "memory");       // stores -> loads (same wave)

        // ---- P as A-fragment (16 rows x 32 keys) ----
        v16h pa;
        {
            v8h lo = *(const v8h*)&lds[warp][n][16 * h];
            v8h hi = *(const v8h*)&lds[warp][n][16 * h + 8];
#pragma unroll
            for (int i = 0; i < 8; ++i) { pa[i] = lo[i]; pa[i + 8] = hi[i]; }
        }
        asm volatile("s_wait_dscnt 0" ::: "memory");       // loads done before next overwrite

        // ---- O += P * V  (V stored d-major so B-fragments are contiguous) ----
        v16h vb0 = *(const v16h*)(Vbase + (size_t)(n)      * N_TOK + kt + 16 * h);
        v16h vb1 = *(const v16h*)(Vbase + (size_t)(16 + n) * N_TOK + kt + 16 * h);
        v16h vb2 = *(const v16h*)(Vbase + (size_t)(32 + n) * N_TOK + kt + 16 * h);
        v16h vb3 = *(const v16h*)(Vbase + (size_t)(48 + n) * N_TOK + kt + 16 * h);
        o0 = wmma_f16(pa, vb0, o0);
        o1 = wmma_f16(pa, vb1, o1);
        o2 = wmma_f16(pa, vb2, o2);
        o3 = wmma_f16(pa, vb3, o3);
    }

    // ---- normalize and store merged-head layout (b, n, c) f16 ----
#pragma unroll
    for (int r = 0; r < 8; ++r) {
        const float inv = 1.0f / rsum[r];
        const int tok = m0 + r + 8 * h;
        _Float16* arow = Att + ((size_t)bq * N_TOK + tok) * CH + head * HD;
        arow[n]      = (_Float16)(o0[r] * inv);
        arow[16 + n] = (_Float16)(o1[r] * inv);
        arow[32 + n] = (_Float16)(o2[r] * inv);
        arow[48 + n] = (_Float16)(o3[r] * inv);
    }
}

// ---------------------------------------------------------------------------
// Kernel 3: output projection + transpose back to (B, C, H, W) f32.
// ---------------------------------------------------------------------------
__global__ __launch_bounds__(256) void out_proj_kernel(
    const _Float16* __restrict__ Att, const float* __restrict__ w_o,
    const float* __restrict__ b_o, float* __restrict__ out) {
    const int lane = threadIdx.x & 31;
    const int warp = threadIdx.x >> 5;
    const int wid  = blockIdx.x * 8 + warp;

    const int col_tile = wid & 15;        // 256/16 output-feature tiles
    const int row_tile = wid >> 4;        // 1024 token tiles
    const int b  = row_tile >> 8;
    const int m0 = (row_tile & 255) * 16;

    const int h  = lane >> 4;
    const int nn = lane & 15;
    const int t  = m0 + nn;
    const int o  = col_tile * 16 + nn;

    const _Float16* arow = Att + ((size_t)b * N_TOK + t) * CH;

    v8f acc = {};
#pragma unroll
    for (int kc = 0; kc < 8; ++kc) {
        v16h a = load_a_f16(arow + kc * 32, h);           // contiguous f16
        const float* wr = w_o + (size_t)o * CH + kc * 32 + 16 * h;
        v16h bf;
#pragma unroll
        for (int i = 0; i < 16; ++i) bf[i] = (_Float16)wr[i];
        acc = wmma_f16(a, bf, acc);
    }

    const float bias = b_o[o];
#pragma unroll
    for (int r = 0; r < 8; ++r) {
        const int tok = m0 + r + 8 * h;
        out[((size_t)b * CH + o) * N_TOK + tok] = acc[r] + bias;  // (B,C,H*W)
    }
}

// ---------------------------------------------------------------------------
extern "C" void kernel_launch(void* const* d_in, const int* in_sizes, int n_in,
                              void* d_out, int out_size, void* d_ws, size_t ws_size,
                              hipStream_t stream) {
    const float* x     = (const float*)d_in[0];
    const float* w_qkv = (const float*)d_in[1];
    const float* b_qkv = (const float*)d_in[2];
    const float* w_o   = (const float*)d_in[3];
    const float* b_o   = (const float*)d_in[4];
    float* out         = (float*)d_out;

    const size_t elems = (size_t)4 * NH * N_TOK * HD;     // 4,194,304 per buffer
    _Float16* Qb  = (_Float16*)d_ws;
    _Float16* Kb  = Qb + elems;
    _Float16* Vt  = Kb + elems;
    _Float16* Att = Vt + elems;                           // (B, N, C) f16

    // 1) QKV projection: 1024 row tiles x 48 col tiles = 49152 waves
    qkv_proj_kernel<<<6144, 256, 0, stream>>>(x, w_qkv, b_qkv, Qb, Kb, Vt);
    // 2) flash attention: 4 * 4 * 256 = 4096 waves
    flash_attn_kernel<<<512, 256, 0, stream>>>(Qb, Kb, Vt, Att);
    // 3) output projection: 1024 x 16 = 16384 waves
    out_proj_kernel<<<2048, 256, 0, stream>>>(Att, w_o, b_o, out);
}
